// ContextualLoss_forward_4243427689207
// MI455X (gfx1250) — matrix operand order repaired and checked by reading, not compiled, not run on hardware
//
#include <hip/hip_runtime.h>
#include <hip/hip_bf16.h>

typedef _Float16 v16h __attribute__((ext_vector_type(16)));
typedef _Float16 v8h  __attribute__((ext_vector_type(8)));
typedef float    v8f  __attribute__((ext_vector_type(8)));

#define CL_B    4
#define CL_C    256
#define CL_N    4096
#define CL_EPS  2.220446049250313e-16f
#define CL_DEPS 1e-5f
#define CL_HINV 10.0f   // 1 / bandwidth h

#define CL_NSTRIP (CL_N / 16)      // 256 column strips of 16
#define CL_STRIPH (16 * CL_C)      // halves per strip (8 KB)

// ---- Tensor Data Mover availability (this toolchain: 6-arg builtin) -------
#if defined(__has_builtin)
#if __has_builtin(__builtin_amdgcn_tensor_load_to_lds) && \
    __has_builtin(__builtin_amdgcn_s_wait_tensorcnt)
#define HAVE_TDM 1
#endif
#endif
#ifndef HAVE_TDM
#define HAVE_TDM 0
#endif

#if HAVE_TDM
typedef unsigned int u32x4 __attribute__((ext_vector_type(4)));
typedef int          i32x8 __attribute__((ext_vector_type(8)));
typedef int          i32x4 __attribute__((ext_vector_type(4)));

// Issue a TDM load of one contiguous 8 KB strip (1024 x 8-byte elements)
// from global memory into LDS.  D# layout per CDNA5 ISA section 8.3/8.4.
__device__ __forceinline__ void tdm_load_strip(const _Float16* gsrc, _Float16* ldst)
{
  const unsigned long long ga = (unsigned long long)(size_t)gsrc;
  const unsigned int       la = (unsigned int)(size_t)ldst;  // LDS byte offset

  u32x4 g0;
  g0[0] = 1u;                                   // count=1 (valid), user mode
  g0[1] = la;                                   // lds_addr
  g0[2] = (unsigned int)ga;                     // global_addr[31:0]
  g0[3] = (unsigned int)((ga >> 32) & 0x01FFFFFFu) | (2u << 30); // [56:32], type=2

  i32x8 g1;
  g1[0] = (int)(3u << 16);        // wg_mask=0, data_size=3 (8B), no flags
  g1[1] = (int)(1024u << 16);     // abar_addr=0 | tensor_dim0[15:0]=1024
  g1[2] = (int)(1u << 16);        // tensor_dim0[31:16]=0 | tensor_dim1[15:0]=1
  g1[3] = (int)(1024u << 16);     // tensor_dim1[31:16]=0 | tile_dim0=1024
  g1[4] = 1;                      // tile_dim1=1, tile_dim2=0
  g1[5] = 1024;                   // tensor_dim0_stride[31:0]
  g1[6] = (int)(1024u << 16);     // dim0_stride[47:32]=0 | dim1_stride[15:0]=1024
  g1[7] = 0;                      // dim1_stride[47:16]=0

  const i32x4 z4 = {0, 0, 0, 0};              // groups 2/3 unused (<=2D tensor)
  const i32x8 z8 = {0, 0, 0, 0, 0, 0, 0, 0};  // extra operand (clang-23 form)
  __builtin_amdgcn_tensor_load_to_lds(g0, g1, z4, z4, z8, 0);
}
#endif

// ---- wave32 cross-lane reductions over the 16-lane half-groups -------------
// ds_swizzle group-of-32 encoding: offset = (xor_mask<<10) | and_mask(0x1f)
__device__ __forceinline__ float half16_min(float x) {
  x = fminf(x, __int_as_float(__builtin_amdgcn_ds_swizzle(__float_as_int(x), 0x041f)));
  x = fminf(x, __int_as_float(__builtin_amdgcn_ds_swizzle(__float_as_int(x), 0x081f)));
  x = fminf(x, __int_as_float(__builtin_amdgcn_ds_swizzle(__float_as_int(x), 0x101f)));
  x = fminf(x, __int_as_float(__builtin_amdgcn_ds_swizzle(__float_as_int(x), 0x201f)));
  return x;
}
__device__ __forceinline__ float half16_sum(float x) {
  x += __int_as_float(__builtin_amdgcn_ds_swizzle(__float_as_int(x), 0x041f));
  x += __int_as_float(__builtin_amdgcn_ds_swizzle(__float_as_int(x), 0x081f));
  x += __int_as_float(__builtin_amdgcn_ds_swizzle(__float_as_int(x), 0x101f));
  x += __int_as_float(__builtin_amdgcn_ds_swizzle(__float_as_int(x), 0x201f));
  return x;
}

// ---- kernel 1: mu_Y[b,c] = mean over spatial of Y; also zero CX accumulators
__global__ __launch_bounds__(256) void cx_mu_kernel(
    const float* __restrict__ Y, float* __restrict__ mu, float* __restrict__ CXs)
{
  __shared__ float sbuf[256];
  const int bc  = blockIdx.x;          // b*C + c
  const int tid = threadIdx.x;
  const float* yp = Y + (size_t)bc * CL_N;
  float s = 0.f;
  for (int n = tid; n < CL_N; n += 256) s += yp[n];
  sbuf[tid] = s;
  __syncthreads();
  for (int off = 128; off > 0; off >>= 1) {
    if (tid < off) sbuf[tid] += sbuf[tid + off];
    __syncthreads();
  }
  if (tid == 0) mu[bc] = sbuf[0] * (1.0f / CL_N);
  if (blockIdx.x == 0 && tid < CL_B) CXs[tid] = 0.f;
}

// ---- kernel 2: center, L2-normalize over C, convert to f16, transpose to [B][N][C]
__global__ __launch_bounds__(256) void cx_norm_kernel(
    const float* __restrict__ X, const float* __restrict__ Y,
    const float* __restrict__ mu,
    _Float16* __restrict__ Xt, _Float16* __restrict__ Yt)
{
  const int g = blockIdx.x * 256 + threadIdx.x;   // g = b*N + n
  const int b = g >> 12;
  const int n = g & (CL_N - 1);
  const float* xb  = X + (size_t)b * CL_C * CL_N + n;
  const float* yb  = Y + (size_t)b * CL_C * CL_N + n;
  const float* mub = mu + b * CL_C;

  float sx = 0.f, sy = 0.f;
  for (int c = 0; c < CL_C; ++c) {
    float m = mub[c];
    float x = xb[(size_t)c * CL_N] - m;
    float y = yb[(size_t)c * CL_N] - m;
    sx += x * x;
    sy += y * y;
  }
  const float rx = 1.0f / (sqrtf(sx) + CL_EPS);
  const float ry = 1.0f / (sqrtf(sy) + CL_EPS);

  _Float16* xo = Xt + (size_t)g * CL_C;
  _Float16* yo = Yt + (size_t)g * CL_C;
  for (int c = 0; c < CL_C; ++c) {
    float m = mub[c];
    xo[c] = (_Float16)((xb[(size_t)c * CL_N] - m) * rx);
    yo[c] = (_Float16)((yb[(size_t)c * CL_N] - m) * ry);
  }
}

// ---- sweep over all 256 column strips, B tile broadcast through LDS --------
// PASS2=false: stat[] accumulates row minima of d = max(1-s, 0)
// PASS2=true : stat[] accumulates row sums of w = exp(HINV - d*invd)
template <bool PASS2>
__device__ __forceinline__ void cx_sweep(
    const v16h a[8], const _Float16* __restrict__ Yb, _Float16* sB,
    float stat[8], const float invd[8],
    int tid, int wave, int ln16, int hi)
{
  __syncthreads();                                  // sB free for re-staging
#if HAVE_TDM
  if (wave == 0) {                                  // prologue: strip 0 -> buf 0
    tdm_load_strip(Yb, sB);
    __builtin_amdgcn_s_wait_tensorcnt(0);
  }
  __syncthreads();
#else
  {                                                 // prologue: strip 0 -> buf 0
    const uint4* g = (const uint4*)Yb + tid * 2;
    uint4 r0 = g[0], r1 = g[1];
    uint4* l = (uint4*)sB + tid * 2;
    l[0] = r0; l[1] = r1;
  }
  __syncthreads();
#endif

  for (int s = 0; s < CL_NSTRIP; ++s) {
    const bool pre = (s + 1 < CL_NSTRIP);
#if HAVE_TDM
    if (wave == 0 && pre)                           // DMA next strip during compute
      tdm_load_strip(Yb + (size_t)(s + 1) * CL_STRIPH,
                     sB + ((s + 1) & 1) * CL_STRIPH);
#else
    uint4 r0, r1;
    if (pre) {
      const uint4* g = (const uint4*)(Yb + (size_t)(s + 1) * CL_STRIPH) + tid * 2;
      r0 = g[0]; r1 = g[1];
    }
#endif

    // ---- compute on current buffer (uniform across block; EXEC all ones) ----
    // B fragments pipelined one step ahead so each WMMA waits on dscnt<=2
    // (one fragment in flight) instead of draining the LDS queue.
    const _Float16* lb = sB + (s & 1) * CL_STRIPH + ln16 * CL_C + hi * 16;
    v8f acc = {};
    v16h bf = *(const v16h*)lb;
#pragma unroll
    for (int ks = 0; ks < 8; ++ks) {
      v16h nxt;
      if (ks < 7) nxt = *(const v16h*)(lb + (ks + 1) * 32);
      else        nxt = bf;
      acc = __builtin_amdgcn_wmma_f32_16x16x32_f16(
          false, a[ks], false, bf, (short)0, acc, false, false);
      bf = nxt;
    }
#pragma unroll
    for (int v = 0; v < 8; ++v) {
      float d = fmaxf(1.0f - acc[v], 0.0f);         // row = wave*16 + v + 8*hi
      if (PASS2) stat[v] += __expf(CL_HINV - d * invd[v]);
      else       stat[v] = fminf(stat[v], d);
    }

#if HAVE_TDM
    if (wave == 0) __builtin_amdgcn_s_wait_tensorcnt(0);
#else
    if (pre) {
      uint4* l = (uint4*)(sB + ((s + 1) & 1) * CL_STRIPH) + tid * 2;
      l[0] = r0; l[1] = r1;
    }
#endif
    __syncthreads();
  }
}

// ---- kernel 3: 128 rows per WG (8 waves x 16 rows), B strips broadcast via LDS
__global__ __launch_bounds__(256) void cx_gemm_kernel(
    const _Float16* __restrict__ Xt, const _Float16* __restrict__ Yt,
    float* __restrict__ CXs)
{
  __shared__ _Float16 sB[2 * CL_STRIPH];            // 2 x 8 KB double buffer

  const int blk  = blockIdx.x;
  const int b    = blk >> 5;                        // N/128 = 32 row blocks/batch
  const int i0   = (blk & 31) << 7;
  const int tid  = threadIdx.x;
  const int wave = tid >> 5;
  const int lane = tid & 31;
  const int ln16 = lane & 15;
  const int hi   = lane >> 4;

  // A tile (16 rows x 256 K) resident in registers; A/B k order used here (not ISA table order):
  // lane m<16 holds M=m; halves 0..15 -> K = k0 + hi*16 + h (contiguous).
  const _Float16* arow =
      Xt + (size_t)(b * CL_N + i0 + wave * 16 + ln16) * CL_C;
  v16h a[8];
#pragma unroll
  for (int ks = 0; ks < 8; ++ks) {
    const int k0 = ks * 32 + hi * 16;
    v8h a0 = *(const v8h*)(arow + k0);
    v8h a1 = *(const v8h*)(arow + k0 + 8);
    a[ks] = __builtin_shufflevector(a0, a1,
            0,1,2,3,4,5,6,7,8,9,10,11,12,13,14,15);
  }

  const _Float16* Yb = Yt + (size_t)b * CL_N * CL_C;  // strips are contiguous 8 KB

  // -------- pass 1: row minima of d --------
  float dmin[8];
#pragma unroll
  for (int v = 0; v < 8; ++v) dmin[v] = 3.0e38f;
  cx_sweep<false>(a, Yb, sB, dmin, nullptr, tid, wave, ln16, hi);
#pragma unroll
  for (int v = 0; v < 8; ++v) dmin[v] = half16_min(dmin[v]);

  float invd[8];
#pragma unroll
  for (int v = 0; v < 8; ++v) invd[v] = CL_HINV / (dmin[v] + CL_DEPS);

  // -------- pass 2: row sums of w = exp((1 - d/dmin')/h) --------
  float wsum[8];
#pragma unroll
  for (int v = 0; v < 8; ++v) wsum[v] = 0.0f;
  cx_sweep<true>(a, Yb, sB, wsum, invd, tid, wave, ln16, hi);
#pragma unroll
  for (int v = 0; v < 8; ++v) wsum[v] = half16_sum(wsum[v]);

  // rowCX = w_max / sum_w ; one atomic per 16-lane half-group
  if (ln16 == 0) {
    float part = 0.f;
#pragma unroll
    for (int v = 0; v < 8; ++v)
      part += __expf(CL_HINV * CL_DEPS / (dmin[v] + CL_DEPS)) / wsum[v];
    atomicAdd(&CXs[b], part);
  }
}

// ---- kernel 4: loss = mean_b( -log(CX_b / N) ) ----
__global__ void cx_final_kernel(const float* __restrict__ CXs, float* __restrict__ out)
{
  if (threadIdx.x == 0 && blockIdx.x == 0) {
    float acc = 0.f;
    for (int b = 0; b < CL_B; ++b)
      acc += -__logf(CXs[b] * (1.0f / CL_N));
    out[0] = acc * (1.0f / CL_B);
  }
}

extern "C" void kernel_launch(void* const* d_in, const int* in_sizes, int n_in,
                              void* d_out, int out_size, void* d_ws, size_t ws_size,
                              hipStream_t stream)
{
  const float* X = (const float*)d_in[0];
  const float* Y = (const float*)d_in[1];
  float* out = (float*)d_out;

  // workspace layout: [mu 1024 f32][pad][CXs 4 f32][pad][Xt f16][Yt f16]
  char* ws = (char*)d_ws;
  float*    mu  = (float*)ws;                  // 4 KB
  float*    CXs = (float*)(ws + 4096);         // 16 B
  _Float16* Xt  = (_Float16*)(ws + 8192);      // B*N*C f16 = 8 MB (8 KB aligned)
  _Float16* Yt  = Xt + (size_t)CL_B * CL_N * CL_C;

  cx_mu_kernel  <<<CL_B * CL_C,           256, 0, stream>>>(Y, mu, CXs);
  cx_norm_kernel<<<(CL_B * CL_N) / 256,   256, 0, stream>>>(X, Y, mu, Xt, Yt);
  cx_gemm_kernel<<<CL_B * (CL_N / 128),   256, 0, stream>>>(Xt, Yt, CXs);
  cx_final_kernel<<<1, 32, 0, stream>>>(CXs, out);
}
